// SelfAttention_78503412236683
// MI455X (gfx1250) — compile-verified, matrix operand order checked by reading, not run on hardware
//
#include <hip/hip_runtime.h>
#include <hip/hip_bf16.h>
#include <stdint.h>

// ---------------------------------------------------------------------------
// Problem constants (from the reference)
// ---------------------------------------------------------------------------
constexpr int kSeq   = 2048;
constexpr int kBatch = 4;
constexpr int kEmb   = 2048;
constexpr int kHeads = 16;
constexpr int kHd    = 128;           // head dim
constexpr int kRows  = kSeq * kBatch; // 8192 GEMM rows
constexpr size_t kR  = (size_t)kSeq * kBatch * kEmb; // 16,777,216 elements

// ---------------------------------------------------------------------------
// WMMA types / helpers (CDNA5 gfx1250, wave32)
// ---------------------------------------------------------------------------
typedef __attribute__((ext_vector_type(16))) __bf16 v16bf;
typedef __attribute__((ext_vector_type(8)))  float  v8f;

struct Frag { unsigned int u[8]; };   // 8 VGPRs = 16 bf16 elements

__device__ __forceinline__ v16bf frag_bf(Frag f) {
  return __builtin_bit_cast(v16bf, f);
}
__device__ __forceinline__ unsigned short f2bf(float f) {   // RNE fp32->bf16
  unsigned int u = __builtin_bit_cast(unsigned int, f);
  u += 0x7FFFu + ((u >> 16) & 1u);
  return (unsigned short)(u >> 16);
}
__device__ __forceinline__ unsigned int pack2(float lo, float hi) {
  return (unsigned int)f2bf(lo) | ((unsigned int)f2bf(hi) << 16);
}
// K offset inside a 32-wide chunk held by VGPR v of a 16-bit A/B fragment
// (ISA 7.12.2: v0..3 -> 2v+8g, v4..7 -> 16+2(v-4)+8g, g = lane/16)
__device__ __forceinline__ int klo(int v) { return (v < 4) ? 2 * v : (8 + 2 * v); }

__device__ __forceinline__ v8f wmma_bf16(Frag a, Frag b, v8f c) {
  return __builtin_amdgcn_wmma_f32_16x16x32_bf16(
      false, frag_bf(a), false, frag_bf(b), (short)0, c, false, false);
}

// ---------------------------------------------------------------------------
// Tensor Data Mover (guarded; falls back to cooperative loads)
// This toolchain exposes the 6-arg builtin:
//   (uint32x4 g0, int32x8 g1, int32x4 g2, int32x4 g3, int32x8, i32 cpol)
// ---------------------------------------------------------------------------
#if defined(__HIP_DEVICE_COMPILE__) && __has_builtin(__builtin_amdgcn_tensor_load_to_lds) && __has_builtin(__builtin_amdgcn_s_wait_tensorcnt)
#define USE_TDM 1
typedef __attribute__((ext_vector_type(4))) unsigned int v4u;
typedef __attribute__((ext_vector_type(8))) int         v8i;
typedef __attribute__((ext_vector_type(4))) int         v4i;

// 2D tile load: nrows rows of rowlen bf16 elements, global row stride
// row_stride elements; LDS gets +1 DWORD pad after every (2<<pad_code) DWORDs.
__device__ __forceinline__ void tdm_load_2d(unsigned int lds_byte_off,
                                            const void* gptr,
                                            int rowlen, int nrows,
                                            unsigned long long row_stride,
                                            int pad_code) {
  unsigned long long ga = (unsigned long long)(uintptr_t)gptr;
  v4u g0;
  g0[0] = 1u;                                              // count=1, user mode
  g0[1] = lds_byte_off;                                    // lds_addr
  g0[2] = (unsigned int)(ga & 0xFFFFFFFFull);              // global_addr lo
  g0[3] = (unsigned int)((ga >> 32) & 0x01FFFFFFull) | (2u << 30); // hi | type=2
  v8i g1;
  g1[0] = (int)((1u << 16) |                               // data_size = 2B
                (1u << 20) |                               // pad_enable
                ((unsigned)pad_code << 22));               // pad_interval
  g1[1] = (int)(((unsigned)rowlen & 0xFFFFu) << 16);       // tensor_dim0 lo
  g1[2] = (int)((((unsigned)rowlen >> 16) & 0xFFFFu) |
                (((unsigned)nrows & 0xFFFFu) << 16));      // dim0 hi | dim1 lo
  g1[3] = (int)((((unsigned)nrows >> 16) & 0xFFFFu) |
                (((unsigned)rowlen & 0xFFFFu) << 16));     // dim1 hi | tile_dim0
  g1[4] = (int)((unsigned)nrows & 0xFFFFu);                // tile_dim1
  g1[5] = (int)(unsigned)(row_stride & 0xFFFFFFFFull);     // dim0_stride lo
  g1[6] = (int)(unsigned)((row_stride >> 32) & 0xFFFFull); // dim0_stride hi
  g1[7] = 0;
  v4i z4 = {0, 0, 0, 0};
  v8i z8 = {0, 0, 0, 0, 0, 0, 0, 0};
  __builtin_amdgcn_tensor_load_to_lds(g0, g1, z4, z4, z8, 0);
}
__device__ __forceinline__ void tdm_wait() {
  __builtin_amdgcn_s_wait_tensorcnt((short)0);
}
__device__ __forceinline__ unsigned int lds_off(const void* p) {
  return (unsigned int)(uintptr_t)p;   // generic LDS addr: low 32 bits = offset
}
#endif

// ---------------------------------------------------------------------------
// GEMM:  C[M,N] = A[M,K] * W[N,K]^T (+bias)
//   mode 0: A fp32 (=X); epilogue: +bias, Q->bf16, K/V->kv_cache(fp32)
//           plus Kbf [b,h,t,d] and VTbf [b,h,d,t] bf16 staging
//   mode 1: A bf16 (=ctx); epilogue: fp32 output, no bias
// Block 256x128, BK=32; 8 waves of 64x64 (4x4 WMMA frags each);
// global loads software-pipelined into registers.
// ---------------------------------------------------------------------------
constexpr int BM = 256, BN = 128, BK = 32;
constexpr int LDT = BK + 2;   // padded bf16 stride in LDS

__global__ __launch_bounds__(256) void gemm_bf16_wmma(
    const float* __restrict__ A,
    const unsigned short* __restrict__ Abf,
    const float* __restrict__ W,
    const float* __restrict__ bias,
    float* __restrict__ outF,
    unsigned short* __restrict__ Qbf,
    float* __restrict__ KV,
    unsigned short* __restrict__ Kbf,
    unsigned short* __restrict__ VTbf,
    int M, int N, int K, int mode)
{
  __shared__ unsigned short sA[BM * LDT];   // 17408 B
  __shared__ unsigned short sW[BN * LDT];   //  8704 B

  const int tid  = threadIdx.x;
  const int wv   = tid >> 5;
  const int lane = tid & 31;
  const int ln   = lane & 15;
  const int g8   = (lane >> 4) * 8;
  const int wm   = wv & 3;     // 4 wave-rows (64 rows each)
  const int wn   = wv >> 2;    // 2 wave-cols (64 cols each)
  const int m0   = blockIdx.y * BM;
  const int n0   = blockIdx.x * BN;

  v8f acc[4][4];
#pragma unroll
  for (int mi = 0; mi < 4; ++mi)
#pragma unroll
    for (int ni = 0; ni < 4; ++ni) { v8f z = {}; acc[mi][ni] = z; }

  float4 ra[8];
  uint4  rabf[4];
  float4 rw[4];

  auto preload = [&](int kk) {
    if (mode == 0) {
#pragma unroll
      for (int i2 = 0; i2 < 8; ++i2) {
        int idx = tid + i2 * 256;          // 2048 float4: 256 rows x 8
        int row = idx >> 3, kq = (idx & 7) * 4;
        ra[i2] = *(const float4*)&A[(size_t)(m0 + row) * K + kk + kq];
      }
    } else {
#pragma unroll
      for (int i2 = 0; i2 < 4; ++i2) {
        int idx = tid + i2 * 256;          // 1024 uint4: 256 rows x 4
        int row = idx >> 2, kq = (idx & 3) * 8;
        rabf[i2] = *(const uint4*)&Abf[(size_t)(m0 + row) * K + kk + kq];
      }
    }
#pragma unroll
    for (int i2 = 0; i2 < 4; ++i2) {
      int idx = tid + i2 * 256;            // 1024 float4: 128 rows x 8
      int row = idx >> 3, kq = (idx & 7) * 4;
      rw[i2] = *(const float4*)&W[(size_t)(n0 + row) * K + kk + kq];
    }
  };

  auto commit = [&]() {
    if (mode == 0) {
#pragma unroll
      for (int i2 = 0; i2 < 8; ++i2) {
        int idx = tid + i2 * 256;
        int row = idx >> 3, kq = (idx & 7) * 4;
        unsigned int* d = (unsigned int*)&sA[row * LDT + kq];
        d[0] = pack2(ra[i2].x, ra[i2].y);
        d[1] = pack2(ra[i2].z, ra[i2].w);
      }
    } else {
#pragma unroll
      for (int i2 = 0; i2 < 4; ++i2) {
        int idx = tid + i2 * 256;
        int row = idx >> 2, kq = (idx & 3) * 8;
        unsigned int* d = (unsigned int*)&sA[row * LDT + kq];
        d[0] = rabf[i2].x; d[1] = rabf[i2].y; d[2] = rabf[i2].z; d[3] = rabf[i2].w;
      }
    }
#pragma unroll
    for (int i2 = 0; i2 < 4; ++i2) {
      int idx = tid + i2 * 256;
      int row = idx >> 3, kq = (idx & 7) * 4;
      unsigned int* d = (unsigned int*)&sW[row * LDT + kq];
      d[0] = pack2(rw[i2].x, rw[i2].y);
      d[1] = pack2(rw[i2].z, rw[i2].w);
    }
  };

  preload(0);
  for (int kk = 0; kk < K; kk += BK) {
    __syncthreads();                       // prior compute done with LDS
    commit();
    __syncthreads();
    if (kk + BK < K) preload(kk + BK);     // global loads overlap WMMA phase

    Frag af[4], bfr[4];
#pragma unroll
    for (int mi = 0; mi < 4; ++mi) {
      int row = wm * 64 + mi * 16 + ln;
#pragma unroll
      for (int v = 0; v < 8; ++v)
        af[mi].u[v] = *(const unsigned int*)&sA[row * LDT + klo(v) + g8];
    }
#pragma unroll
    for (int ni = 0; ni < 4; ++ni) {
      int row = wn * 64 + ni * 16 + ln;
#pragma unroll
      for (int v = 0; v < 8; ++v)
        bfr[ni].u[v] = *(const unsigned int*)&sW[row * LDT + klo(v) + g8];
    }
#pragma unroll
    for (int mi = 0; mi < 4; ++mi)
#pragma unroll
      for (int ni = 0; ni < 4; ++ni)
        acc[mi][ni] = wmma_bf16(af[mi], bfr[ni], acc[mi][ni]);
  }

  // ---- epilogue -------------------------------------------------------
#pragma unroll
  for (int mi = 0; mi < 4; ++mi)
#pragma unroll
    for (int ni = 0; ni < 4; ++ni)
#pragma unroll
      for (int v = 0; v < 8; ++v) {
        int m = m0 + wm * 64 + mi * 16 + g8 + v;
        int n = n0 + wn * 64 + ni * 16 + ln;
        float val = acc[mi][ni][v];
        if (mode == 0) {
          val += bias[n];
          int s = m >> 2, bb = m & 3;
          int h = n / (3 * kHd);
          int r = n - h * (3 * kHd);
          if (r < kHd) {
            Qbf[(size_t)m * kEmb + h * kHd + r] = f2bf(val);
          } else {
            int e = (r < 2 * kHd) ? 0 : 1;
            int d = r - kHd - e * kHd;
            KV[(((size_t)e * kSeq + s) * kBatch + bb) * kEmb + h * kHd + d] = val;
            int bh = bb * kHeads + h;
            if (e == 0)
              Kbf[((size_t)bh * kSeq + s) * kHd + d] = f2bf(val);
            else
              VTbf[((size_t)bh * kHd + d) * kSeq + s] = f2bf(val);
          }
        } else {
          outF[(size_t)m * N + n] = val;
        }
      }
}

// ---------------------------------------------------------------------------
// Flash attention: WG = (128 q rows, b, h); 8 waves x 16 rows.
// K/V^T tiles land in LDS via TDM (double-buffered, pipelined) or
// cooperative bf16 loads; S=QK^T and O+=PV via WMMA; fp32 online softmax.
// ---------------------------------------------------------------------------
__global__ __launch_bounds__(256) void attn_flash_wmma(
    const unsigned short* __restrict__ Qbf,   // [SEQ*BATCH, EMB] bf16
    const unsigned short* __restrict__ Kbf,   // [B,H,SEQ,HD]  bf16
    const unsigned short* __restrict__ VTbf,  // [B,H,HD,SEQ]  bf16
    unsigned short* __restrict__ ctxbf)       // [SEQ*BATCH, EMB] bf16
{
  const int qblk = blockIdx.x;                // 0..15 (128 rows each)
  const int bh   = blockIdx.y;                // b*HEADS + h
  const int b    = bh >> 4;
  const int h    = bh & 15;
  const int tid  = threadIdx.x;
  const int wv   = tid >> 5;
  const int lane = tid & 31;
  const int ln   = lane & 15;
  const int g8   = (lane >> 4) * 8;

  __shared__ unsigned short sK [2][64 * 130];   // [key][dim], stride 130
  __shared__ unsigned short sVT[2][128 * 66];   // [dim][key], stride 66
  __shared__ unsigned short sP [8 * 16 * 66];   // per-wave P strip

  // ---- Q fragments (persist in registers) ---------------------------
  Frag qf[4];
  {
    const int qrow = qblk * 128 + wv * 16 + ln;
    const size_t qb = ((size_t)(qrow * kBatch + b)) * kEmb + h * kHd;
#pragma unroll
    for (int c = 0; c < 4; ++c)
#pragma unroll
      for (int v = 0; v < 8; ++v)
        qf[c].u[v] = *(const unsigned int*)&Qbf[qb + c * 32 + klo(v) + g8];
  }

  float m_st[8], l_st[8];
  v8f   oacc[8];
#pragma unroll
  for (int v = 0; v < 8; ++v) { m_st[v] = -INFINITY; l_st[v] = 0.0f; }
#pragma unroll
  for (int dt = 0; dt < 8; ++dt) { v8f z = {}; oacc[dt] = z; }

  const float norm   = 0.08838834764831845f;   // 1/sqrt(128)
  const int   qg     = qblk * 128 + wv * 16 + g8;
  const int   ntiles = 2 * qblk + 2;           // causal: t0 <= q_max
  const unsigned short* kg0 = Kbf  + (size_t)bh * kSeq * kHd;
  const unsigned short* vg0 = VTbf + (size_t)bh * kHd * kSeq;

#if defined(USE_TDM)
  if (wv == 0) {   // prologue: tile 0 into buffer 0
    tdm_load_2d(lds_off(&sK[0][0]),  kg0, kHd, 64, kHd, 5);  // 64 dw + 1 pad
    tdm_load_2d(lds_off(&sVT[0][0]), vg0, 64, kHd, kSeq, 4); // 32 dw + 1 pad
  }
#endif

  for (int it = 0; it < ntiles; ++it) {
    const int buf = it & 1;
    const int t0  = it * 64;
#if defined(USE_TDM)
    if (wv == 0) tdm_wait();                 // current buffer complete
    __syncthreads();                         // publish; prev compute also done
    if (wv == 0 && it + 1 < ntiles) {        // prefetch next tile -> other buf
      tdm_load_2d(lds_off(&sK[buf ^ 1][0]),
                  kg0 + (size_t)(t0 + 64) * kHd, kHd, 64, kHd, 5);
      tdm_load_2d(lds_off(&sVT[buf ^ 1][0]),
                  vg0 + (t0 + 64), 64, kHd, kSeq, 4);
    }
#else
    // cooperative bf16 tile loads
#pragma unroll
    for (int i2 = 0; i2 < 4; ++i2) {         // K: 64 rows x 16 uint4
      int idx = tid + i2 * 256;
      int r = idx >> 4, c = (idx & 15) * 8;
      uint4 u = *(const uint4*)&kg0[((size_t)(t0 + r)) * kHd + c];
      unsigned int* d = (unsigned int*)&sK[buf][r * 130 + c];
      d[0] = u.x; d[1] = u.y; d[2] = u.z; d[3] = u.w;
    }
#pragma unroll
    for (int i2 = 0; i2 < 4; ++i2) {         // V^T: 128 rows x 8 uint4
      int idx = tid + i2 * 256;
      int r = idx >> 3, c = (idx & 7) * 8;
      uint4 u = *(const uint4*)&vg0[(size_t)r * kSeq + t0 + c];
      unsigned int* d = (unsigned int*)&sVT[buf][r * 66 + c];
      d[0] = u.x; d[1] = u.y; d[2] = u.z; d[3] = u.w;
    }
    __syncthreads();
#endif

    // ---- S = Q * K^T (16x64 per wave) -------------------------------
    v8f sacc[4];
#pragma unroll
    for (int nt = 0; nt < 4; ++nt) {
      v8f c = {};
      int krow = nt * 16 + ln;
#pragma unroll
      for (int cc = 0; cc < 4; ++cc) {
        Frag kb;
#pragma unroll
        for (int v = 0; v < 8; ++v)
          kb.u[v] = *(const unsigned int*)&sK[buf][krow * 130 + cc * 32 + klo(v) + g8];
        c = wmma_bf16(qf[cc], kb, c);
      }
      sacc[nt] = c;
    }

    // ---- scale, causal mask, online softmax -------------------------
    float ps[4][8];
#pragma unroll
    for (int v = 0; v < 8; ++v) {
      int q = qg + v;
      float rmax = -INFINITY;
#pragma unroll
      for (int nt = 0; nt < 4; ++nt) {
        int t = t0 + nt * 16 + ln;
        float s = sacc[nt][v] * norm;
        if (t > q) s = -10000.0f;            // reference mask value
        ps[nt][v] = s;
        rmax = fmaxf(rmax, s);
      }
#pragma unroll
      for (int off = 1; off < 16; off <<= 1)
        rmax = fmaxf(rmax, __shfl_xor(rmax, off, 32));
      float mnew  = fmaxf(m_st[v], rmax);
      float alpha = __expf(m_st[v] - mnew);
      m_st[v] = mnew;
      float rsum = 0.0f;
#pragma unroll
      for (int nt = 0; nt < 4; ++nt) {
        float p = __expf(ps[nt][v] - mnew);
        ps[nt][v] = p;
        rsum += p;
      }
#pragma unroll
      for (int off = 1; off < 16; off <<= 1)
        rsum += __shfl_xor(rsum, off, 32);
      l_st[v] = l_st[v] * alpha + rsum;
#pragma unroll
      for (int dt = 0; dt < 8; ++dt) oacc[dt][v] *= alpha;
    }

    // ---- P: D-layout -> bf16 A-layout via per-wave LDS strip --------
    unsigned short* pw = &sP[wv * 16 * 66];
#pragma unroll
    for (int nt = 0; nt < 4; ++nt)
#pragma unroll
      for (int v = 0; v < 8; ++v)
        pw[(v + g8) * 66 + nt * 16 + ln] = f2bf(ps[nt][v]);

    Frag pf[2];
#pragma unroll
    for (int c = 0; c < 2; ++c)
#pragma unroll
      for (int v = 0; v < 8; ++v)
        pf[c].u[v] = *(const unsigned int*)&pw[ln * 66 + c * 32 + klo(v) + g8];

    // ---- O += P * V --------------------------------------------------
#pragma unroll
    for (int dt = 0; dt < 8; ++dt) {
      int drow = dt * 16 + ln;
#pragma unroll
      for (int c = 0; c < 2; ++c) {
        Frag vb;
#pragma unroll
        for (int v = 0; v < 8; ++v)
          vb.u[v] = *(const unsigned int*)&sVT[buf][drow * 66 + c * 32 + klo(v) + g8];
        oacc[dt] = wmma_bf16(pf[c], vb, oacc[dt]);
      }
    }
    __syncthreads();                          // done with this buffer
  }

  // ---- normalize + write ctx (bf16) ----------------------------------
#pragma unroll
  for (int dt = 0; dt < 8; ++dt)
#pragma unroll
    for (int v = 0; v < 8; ++v) {
      int q = qg + v;
      int d = dt * 16 + ln;
      float val = oacc[dt][v] / l_st[v];
      ctxbf[((size_t)(q * kBatch + b)) * kEmb + h * kHd + d] = f2bf(val);
    }
}

// ---------------------------------------------------------------------------
__global__ void copy_bias(const float* __restrict__ src, float* __restrict__ dst) {
  int i = blockIdx.x * blockDim.x + threadIdx.x;
  if (i < kEmb) dst[i] = src[i];
}

// ---------------------------------------------------------------------------
extern "C" void kernel_launch(void* const* d_in, const int* in_sizes, int n_in,
                              void* d_out, int out_size, void* d_ws, size_t ws_size,
                              hipStream_t stream) {
  (void)in_sizes; (void)n_in; (void)out_size; (void)ws_size;
  const float* X    = (const float*)d_in[0];
  // d_in[1] = mask (causal; reproduced analytically)
  const float* Wqkv = (const float*)d_in[2];
  const float* bqkv = (const float*)d_in[3];
  const float* Wd   = (const float*)d_in[4];
  const float* bd   = (const float*)d_in[5];

  float* out = (float*)d_out;                       // [SEQ,BATCH,EMB]
  float* kv  = out + kR;                            // (2,SEQ,BATCH,H,HD) fp32
  float* bdo = kv + 2 * kR;                         // [EMB]

  unsigned short* Qbf   = (unsigned short*)d_ws;    // 32 MB each region
  unsigned short* ctxbf = Qbf + kR;
  unsigned short* Kbf   = ctxbf + kR;
  unsigned short* VTbf  = Kbf + kR;

  // 1) QKV projection + scatter (Q/K bf16, V^T bf16, kv_cache fp32)
  dim3 g1(3 * kEmb / BN, kRows / BM);               // 48 x 32
  gemm_bf16_wmma<<<g1, 256, 0, stream>>>(X, nullptr, Wqkv, bqkv,
                                         nullptr, Qbf, kv, Kbf, VTbf,
                                         kRows, 3 * kEmb, kEmb, 0);

  // 2) causal flash attention (TDM-pipelined tiles)
  dim3 g2(kSeq / 128, kBatch * kHeads);             // 16 x 64
  attn_flash_wmma<<<g2, 256, 0, stream>>>(Qbf, Kbf, VTbf, ctxbf);

  // 3) output projection (reference adds no bias here)
  dim3 g3(kEmb / BN, kRows / BM);                   // 16 x 32
  gemm_bf16_wmma<<<g3, 256, 0, stream>>>(nullptr, ctxbf, Wd, nullptr,
                                         out, nullptr, nullptr, nullptr, nullptr,
                                         kRows, kEmb, kEmb, 1);

  // 4) pass through b_dense
  copy_bias<<<(kEmb + 255) / 256, 256, 0, stream>>>(bd, bdo);
}
